// MultiHeadAttention_75093208203490
// MI455X (gfx1250) — matrix operand order repaired and checked by reading, not compiled, not run on hardware
//
#include <hip/hip_runtime.h>

typedef unsigned int u32;
typedef __attribute__((ext_vector_type(16))) __bf16 v16bf;
typedef __attribute__((ext_vector_type(8)))  float  v8f;
typedef __attribute__((ext_vector_type(4)))  u32    su4;
typedef __attribute__((ext_vector_type(8)))  u32    su8;

union Frag16 { v16bf v; uint4 q[2]; };

// A-fragment (16x32 bf16): lane row = lane%16; K elements are
// [8*half .. 8*half+7] and [16+8*half .. 16+8*half+7] (ISA 7.12.2).
__device__ __forceinline__ v16bf load_frag_a(const __bf16* rowBase, int half) {
  Frag16 f;
  f.q[0] = *(const uint4*)(rowBase + half * 8);
  f.q[1] = *(const uint4*)(rowBase + half * 8 + 16);
  return f.v;
}

// B-fragment (32x16 bf16, column-major source): lane col = lane%16;
// K elements are [8*half .. 8*half+7] and [16+8*half .. 16+8*half+7].
__device__ __forceinline__ v16bf load_frag_b(const __bf16* colBase, int half) {
  Frag16 f;
  f.q[0] = *(const uint4*)(colBase + half * 8);
  f.q[1] = *(const uint4*)(colBase + half * 8 + 16);
  return f.v;
}

__device__ __forceinline__ v8f wmma_bf16(v16bf a, v16bf b, v8f c) {
  return __builtin_amdgcn_wmma_f32_16x16x32_bf16(false, a, false, b, (short)0, c,
                                                 false, false);
}

// ---------------------------------------------------------------- TDM
// Issue a 2D tensor_load_to_lds: tile_h rows x tile_w elems (2-byte elems),
// source rows strided by row_stride elems.  D# layout per CDNA5 ISA ch.8.
__device__ __forceinline__ void tdm_load_tile_2d(u32 lds_addr, const void* gaddr,
                                                 u32 tile_w, u32 tile_h,
                                                 u32 row_stride, u32 tens_rows) {
  union G0 { su4 v; u32 w[4]; } g0 = {};
  union G1 { su8 v; u32 w[8]; } g1 = {};
  su4 z = (su4)(0u);
  unsigned long long ga = (unsigned long long)(uintptr_t)gaddr;
  g0.w[0] = 1u;                                   // count=1 (valid descriptor)
  g0.w[1] = lds_addr;                             // LDS byte address
  g0.w[2] = (u32)ga;                              // global_addr[31:0]
  g0.w[3] = (u32)(ga >> 32) | (2u << 30);         // global_addr[56:32], type=2
  g1.w[0] = 1u << 16;                             // data_size=1 -> 2-byte elems
  g1.w[1] = (row_stride & 0xFFFFu) << 16;         // tensor_dim0[15:0]
  g1.w[2] = (row_stride >> 16) | ((tens_rows & 0xFFFFu) << 16);  // dim0 hi, dim1 lo
  g1.w[3] = (tens_rows >> 16) | (tile_w << 16);   // dim1 hi, tile_dim0
  g1.w[4] = tile_h;                               // tile_dim1 (tile_dim2 = 0)
  g1.w[5] = row_stride;                           // tensor_dim0_stride[31:0]
  g1.w[6] = 0u;
  g1.w[7] = 0u;
  asm volatile("tensor_load_to_lds %0, %1, %2, %3"
               :: "s"(g0.v), "s"(g1.v), "s"(z), "s"(z)
               : "memory");
}

// ---------------------------------------------------------------- conversions
__global__ void cvt_f32_to_bf16(const float* __restrict__ in,
                                __bf16* __restrict__ out, int n) {
  int i = blockIdx.x * blockDim.x + threadIdx.x;
  if (i < n) out[i] = (__bf16)in[i];
}

// W[K,N] fp32 -> Wt[N,K] bf16
__global__ void transpose_to_bf16(const float* __restrict__ W,
                                  __bf16* __restrict__ Wt, int K, int N) {
  int i = blockIdx.x * blockDim.x + threadIdx.x;
  if (i < K * N) {
    int k = i / N, n = i - k * N;
    Wt[n * K + k] = (__bf16)W[i];
  }
}

// ---------------------------------------------------------------- GEMM
// C[M,N] = A[M,K] @ Bt[N,K]^T + bias.  Block = 256 thr = 8 waves (2x4),
// wave tile 32x64, block tile 64x256.  Epilogue modes (template):
//   0: fp32 row-major [M,N]
//   1: bf16 scattered to [B,H,S,D]  (row = b*2048+s, col = h*96+d)
//   2: bf16 scattered to [B,H,D,S]  (transposed V)
template <int MODE>
__global__ __launch_bounds__(256)
void gemm_bf16_wmma(const __bf16* __restrict__ A, const __bf16* __restrict__ Bt,
                    const float* __restrict__ bias, void* __restrict__ out,
                    int M, int N, int K) {
  const int lane = threadIdx.x & 31;
  const int wave = threadIdx.x >> 5;
  const int half = lane >> 4;
  const int l16  = lane & 15;
  const int wm = wave >> 2;                 // 0..1
  const int wn = wave & 3;                  // 0..3
  const int mBase = blockIdx.y * 64 + wm * 32;
  const int nBase = blockIdx.x * 256 + wn * 64;

  v8f acc[2][4];
#pragma unroll
  for (int mt = 0; mt < 2; mt++)
#pragma unroll
    for (int ct = 0; ct < 4; ct++)
      acc[mt][ct] = (v8f){0.f, 0.f, 0.f, 0.f, 0.f, 0.f, 0.f, 0.f};

  for (int kk = 0; kk < K; kk += 32) {
    v16bf a[2], b[4];
#pragma unroll
    for (int mt = 0; mt < 2; mt++)
      a[mt] = load_frag_a(A + (size_t)(mBase + mt * 16 + l16) * K + kk, half);
#pragma unroll
    for (int ct = 0; ct < 4; ct++)
      b[ct] = load_frag_b(Bt + (size_t)(nBase + ct * 16 + l16) * K + kk, half);
#pragma unroll
    for (int mt = 0; mt < 2; mt++)
#pragma unroll
      for (int ct = 0; ct < 4; ct++)
        acc[mt][ct] = wmma_bf16(a[mt], b[ct], acc[mt][ct]);
  }

  // C layout: lanes 0-15: row = r, lanes 16-31: row = 8+r; col = lane%16.
#pragma unroll
  for (int mt = 0; mt < 2; mt++) {
#pragma unroll
    for (int ct = 0; ct < 4; ct++) {
      const int col = nBase + ct * 16 + l16;
      const float bv = bias[col];
#pragma unroll
      for (int r = 0; r < 8; r++) {
        const int row = mBase + mt * 16 + r + 8 * half;
        const float v = acc[mt][ct][r] + bv;
        if (MODE == 0) {
          ((float*)out)[(size_t)row * N + col] = v;
        } else {
          const int bb = row >> 11, s = row & 2047;   // S = 2048
          const int h = col / 96, d = col - h * 96;   // N = 768 here
          __bf16* o = (__bf16*)out;
          if (MODE == 1)
            o[(((size_t)(bb * 8 + h)) * 2048 + s) * 96 + d] = (__bf16)v;
          else
            o[(((size_t)(bb * 8 + h)) * 96 + d) * 2048 + s] = (__bf16)v;
        }
      }
    }
  }
}

// ---------------------------------------------------------------- attention
// Q,K: [B,H,S,D] bf16.  Vt: [B,H,D,S] bf16.  Out: bf16 [B,S,H*D].
// Block = 256 thr = 8 waves; wave owns 16 query rows; grid (S/128, B*H).
// K tiles are double-buffered in LDS via the Tensor Data Mover (TENSORcnt
// pipelining: tile i+1 streams in while tile i is consumed).  Flash-style
// online softmax; reference quirk (softmax, THEN * SCALING) folded into the
// output scale.
__global__ __launch_bounds__(256)
void attn_flash_wmma(const __bf16* __restrict__ Q, const __bf16* __restrict__ Km,
                     const __bf16* __restrict__ Vt, __bf16* __restrict__ outb) {
  constexpr int S = 2048, D = 96;
  constexpr int NT = S / 64;                       // 32 key tiles
  constexpr float SCALING = 0.10206207261596575f;  // 96^-0.5
  __shared__ __align__(16) __bf16 ldsK[2][64][96];   // TDM double buffer
  __shared__ __align__(16) __bf16 lds_p[8][16][72];  // per-wave P tile, padded

  const int lane = threadIdx.x & 31;
  const int wave = threadIdx.x >> 5;
  const int half = lane >> 4;
  const int l16  = lane & 15;
  const int bh = blockIdx.y;                 // 0..15
  const int b = bh >> 3, h = bh & 7;
  const int qBase = blockIdx.x * 128 + wave * 16;

  const __bf16* Qh = Q  + (size_t)bh * S * D;
  const __bf16* Kh = Km + (size_t)bh * S * D;
  const __bf16* Vh = Vt + (size_t)bh * D * S;

  v8f o[6];
#pragma unroll
  for (int dt = 0; dt < 6; dt++)
    o[dt] = (v8f){0.f, 0.f, 0.f, 0.f, 0.f, 0.f, 0.f, 0.f};
  float m[8], l[8];
#pragma unroll
  for (int r = 0; r < 8; r++) { m[r] = -1.0e30f; l[r] = 0.f; }

  // Q fragments are loop-invariant: 3 K-steps cover D = 96.
  v16bf qf[3];
#pragma unroll
  for (int ks = 0; ks < 3; ks++)
    qf[ks] = load_frag_a(Qh + (size_t)(qBase + l16) * D + ks * 32, half);

  // Prologue: stage key tile 0 into buffer 0.
  if (wave == 0)
    tdm_load_tile_2d((u32)(uintptr_t)&ldsK[0][0][0], Kh,
                     /*tile_w=*/96, /*tile_h=*/64,
                     /*row_stride=*/96, /*tens_rows=*/2048);

  for (int it = 0; it < NT; it++) {
    const int kt = it * 64;
    const __bf16(*kbuf)[96] = ldsK[it & 1];

    // Pipeline: issue DMA for tile it+1, then wait for tile it (<=1 in
    // flight; TDM ops from one wave complete in order).
    if (wave == 0) {
      if (it + 1 < NT) {
        tdm_load_tile_2d((u32)(uintptr_t)&ldsK[(it + 1) & 1][0][0],
                         Kh + (size_t)(kt + 64) * D,
                         /*tile_w=*/96, /*tile_h=*/64,
                         /*row_stride=*/96, /*tens_rows=*/2048);
        __builtin_amdgcn_s_wait_tensorcnt(1);
      } else {
        __builtin_amdgcn_s_wait_tensorcnt(0);
      }
    }
    if (kt + 128 < S)  // warm L2 two tiles ahead of the TDM
      __builtin_prefetch(Kh + (size_t)(kt + 128) * D + lane * 64, 0, 1);
    __syncthreads();   // tile it visible to all waves

    // ---- E = Q @ K^T  (16 queries x 64 keys), K from LDS ----
    v8f e[4];
#pragma unroll
    for (int ct = 0; ct < 4; ct++) {
      e[ct] = (v8f){0.f, 0.f, 0.f, 0.f, 0.f, 0.f, 0.f, 0.f};
#pragma unroll
      for (int ks = 0; ks < 3; ks++) {
        v16bf kfrag = load_frag_b(&kbuf[ct * 16 + l16][ks * 32], half);
        e[ct] = wmma_bf16(qf[ks], kfrag, e[ct]);
      }
    }

    // ---- online softmax (rows live in 16-lane halves) ----
    float tmax[8];
#pragma unroll
    for (int r = 0; r < 8; r++) {
      float v = e[0][r];
      v = fmaxf(v, e[1][r]); v = fmaxf(v, e[2][r]); v = fmaxf(v, e[3][r]);
#pragma unroll
      for (int mask = 1; mask <= 8; mask <<= 1)
        v = fmaxf(v, __shfl_xor(v, mask, 32));
      tmax[r] = v;
    }
    float alpha[8];
#pragma unroll
    for (int r = 0; r < 8; r++) {
      float mn = fmaxf(m[r], tmax[r]);
      alpha[r] = __expf(m[r] - mn);
      m[r] = mn;
    }
    float rs[8];
#pragma unroll
    for (int r = 0; r < 8; r++) rs[r] = 0.f;
#pragma unroll
    for (int ct = 0; ct < 4; ct++)
#pragma unroll
      for (int r = 0; r < 8; r++) {
        float p = __expf(e[ct][r] - m[r]);
        e[ct][r] = p;
        rs[r] += p;
      }
#pragma unroll
    for (int r = 0; r < 8; r++) {
#pragma unroll
      for (int mask = 1; mask <= 8; mask <<= 1)
        rs[r] += __shfl_xor(rs[r], mask, 32);
      l[r] = l[r] * alpha[r] + rs[r];
    }
#pragma unroll
    for (int dt = 0; dt < 6; dt++)
#pragma unroll
      for (int r = 0; r < 8; r++) o[dt][r] *= alpha[r];

    // ---- P (C-layout) -> LDS -> A-layout fragments ----
#pragma unroll
    for (int ct = 0; ct < 4; ct++)
#pragma unroll
      for (int r = 0; r < 8; r++)
        lds_p[wave][r + 8 * half][ct * 16 + l16] = (__bf16)e[ct][r];
    asm volatile("s_wait_dscnt 0x0" ::: "memory");

    // ---- O += P @ V  (64-key K-dim = 2 WMMA K-steps) ----
#pragma unroll
    for (int ks2 = 0; ks2 < 2; ks2++) {
      v16bf pf = load_frag_a(&lds_p[wave][l16][ks2 * 32], half);
#pragma unroll
      for (int dt = 0; dt < 6; dt++) {
        v16bf vf =
            load_frag_b(Vh + (size_t)(dt * 16 + l16) * S + kt + ks2 * 32, half);
        o[dt] = wmma_bf16(pf, vf, o[dt]);
      }
    }
    asm volatile("s_wait_dscnt 0x0" ::: "memory");  // WAR before next P store
    // All waves done with buffer it&1 before iteration it+1 issues the
    // tile-(it+2) TDM that overwrites it.
    __syncthreads();
  }

  // ---- epilogue: (softmax @ V) * SCALING, scatter to [B,S,H*D] bf16 ----
#pragma unroll
  for (int r = 0; r < 8; r++) {
    const float sc = SCALING / l[r];
    const int s = qBase + r + 8 * half;
#pragma unroll
    for (int dt = 0; dt < 6; dt++) {
      const int col = h * 96 + dt * 16 + l16;
      outb[((size_t)(b * 2048 + s)) * 768 + col] = (__bf16)(o[dt][r] * sc);
    }
  }
}

// ---------------------------------------------------------------- launch
extern "C" void kernel_launch(void* const* d_in, const int* in_sizes, int n_in,
                              void* d_out, int out_size, void* d_ws,
                              size_t ws_size, hipStream_t stream) {
  (void)in_sizes; (void)n_in; (void)out_size; (void)ws_size;
  constexpr int M = 4096, E = 768, S = 2048;
  const float* x  = (const float*)d_in[0];
  const float* Wq = (const float*)d_in[1];
  const float* bq = (const float*)d_in[2];
  const float* Wk = (const float*)d_in[3];
  const float* bk = (const float*)d_in[4];
  const float* Wv = (const float*)d_in[5];
  const float* bv = (const float*)d_in[6];
  const float* Wo = (const float*)d_in[7];
  const float* bo = (const float*)d_in[8];

  char* ws = (char*)d_ws;
  constexpr size_t SZ_XB  = (size_t)M * E * 2;         // x in bf16
  constexpr size_t SZ_WT  = (size_t)E * E * 2;         // one transposed weight
  constexpr size_t SZ_QKV = (size_t)2 * 8 * S * 96 * 2;
  __bf16* xb   = (__bf16*)(ws);
  __bf16* Wqt  = (__bf16*)(ws + SZ_XB);
  __bf16* Wkt  = (__bf16*)(ws + SZ_XB + SZ_WT);
  __bf16* Wvt  = (__bf16*)(ws + SZ_XB + 2 * SZ_WT);
  __bf16* Wot  = (__bf16*)(ws + SZ_XB + 3 * SZ_WT);
  __bf16* Qd   = (__bf16*)(ws + SZ_XB + 4 * SZ_WT);
  __bf16* Kd   = (__bf16*)(ws + SZ_XB + 4 * SZ_WT + SZ_QKV);
  __bf16* Vtd  = (__bf16*)(ws + SZ_XB + 4 * SZ_WT + 2 * SZ_QKV);
  __bf16* attn = (__bf16*)(ws + SZ_XB + 4 * SZ_WT + 3 * SZ_QKV);

  // 1) convert x, transpose+convert weights
  {
    int n = M * E;
    cvt_f32_to_bf16<<<(n + 255) / 256, 256, 0, stream>>>(x, xb, n);
    int nw = E * E;
    int g = (nw + 255) / 256;
    transpose_to_bf16<<<g, 256, 0, stream>>>(Wq, Wqt, E, E);
    transpose_to_bf16<<<g, 256, 0, stream>>>(Wk, Wkt, E, E);
    transpose_to_bf16<<<g, 256, 0, stream>>>(Wv, Wvt, E, E);
    transpose_to_bf16<<<g, 256, 0, stream>>>(Wo, Wot, E, E);
  }

  // 2) QKV projections (WMMA GEMMs, scattered epilogues)
  dim3 gg(E / 256, M / 64);
  gemm_bf16_wmma<1><<<gg, 256, 0, stream>>>(xb, Wqt, bq, Qd,  M, E, E);
  gemm_bf16_wmma<1><<<gg, 256, 0, stream>>>(xb, Wkt, bk, Kd,  M, E, E);
  gemm_bf16_wmma<2><<<gg, 256, 0, stream>>>(xb, Wvt, bv, Vtd, M, E, E);

  // 3) flash attention (double-buffered TDM K tiles, online softmax)
  attn_flash_wmma<<<dim3(S / 128, 16), 256, 0, stream>>>(Qd, Kd, Vtd, attn);

  // 4) output projection -> fp32 d_out
  gemm_bf16_wmma<0><<<gg, 256, 0, stream>>>(attn, Wot, bo, (float*)d_out,
                                            M, E, E);
}
